// CharRNN_55877524521100
// MI455X (gfx1250) — matrix operand order repaired and checked by reading, not compiled, not run on hardware
//
#include <hip/hip_runtime.h>
#include <hip/hip_bf16.h>

// CharRNN fused forward for MI455X (gfx1250, wave32, WMMA bf16 16x16x32).
// One workgroup owns 16 batch rows and runs embed -> 3 RNN layers -> decoder
// fully fused over the T=1024 time loop. All weights resident in LDS (bf16),
// per-wave weight B-fragments hoisted into VGPRs for the whole time loop,
// double-buffered hidden state (1 barrier/layer/step), hardware v_tanh_f32,
// and prefetched token ids.

#define VOCAB   62
#define HID     100
#define NLAYERS 3
#define BATCH   512
#define SEQ     1024

#define WSTRIDE 136   // padded LDS row stride (u16): 272B = 68 dwords -> bank spread
#define NROWS   112   // weight rows padded to 7 tiles of 16
#define MROWS   16    // batch rows per workgroup
#define WAVES   7     // one hidden N-tile (16 cols) per wave: 7*16 = 112 >= 100
#define TPB     (WAVES * 32)

typedef __attribute__((ext_vector_type(16))) __bf16 v16bf;
typedef __attribute__((ext_vector_type(8)))  float  v8f;

__device__ __forceinline__ unsigned short f2bf(float f) {
  unsigned u = __builtin_bit_cast(unsigned, f);
  u += 0x7fffu + ((u >> 16) & 1u);            // round-to-nearest-even
  return (unsigned short)(u >> 16);
}

#if __has_builtin(__builtin_amdgcn_tanhf)
__device__ __forceinline__ float fast_tanh(float x) { return __builtin_amdgcn_tanhf(x); }
#elif __has_builtin(__builtin_amdgcn_tanh_f32)
__device__ __forceinline__ float fast_tanh(float x) { return __builtin_amdgcn_tanh_f32(x); }
#else
__device__ __forceinline__ float fast_tanh(float x) { return tanhf(x); }
#endif

// B fragment (32x16 K x N, bf16): frag[e] = W[row][kbase + e] (contiguous),
// row = tile*16 + (lane&15), kbase = ks*32 + (lane>=16 ? 16 : 0).
__device__ __forceinline__ v16bf load_bfrag(const unsigned short* w, int row, int kbase) {
  const unsigned short* p = w + row * WSTRIDE + kbase;
  v16bf r;
#pragma unroll
  for (int e = 0; e < 16; ++e) r[e] = __builtin_bit_cast(__bf16, p[e]);
  return r;
}

// A fragment (16x32 M x K, bf16), contiguous k order, same as load_bfrag:
// lane row = lane&15; k(e) = kbase + e + 16*hi
__device__ __forceinline__ v16bf load_afrag(const unsigned short* rowp, int kbase, int hi) {
  v16bf r;
#pragma unroll
  for (int e = 0; e < 16; ++e) {
    int k = kbase + e + hi * 16;
    r[e] = __builtin_bit_cast(__bf16, rowp[k]);
  }
  return r;
}

__device__ __forceinline__ v8f wmma_bf16(v16bf a, v16bf b, v8f c) {
  return __builtin_amdgcn_wmma_f32_16x16x32_bf16(false, a, false, b, (short)0, c, false, false);
}

extern "C" __global__ __launch_bounds__(TPB, 1)
void charrnn_fused(const int* __restrict__ ids,
                   const float* __restrict__ emb,
                   const float* __restrict__ W_ih,
                   const float* __restrict__ W_hh,
                   const float* __restrict__ b_ih,
                   const float* __restrict__ b_hh,
                   const float* __restrict__ W_dec,
                   const float* __restrict__ b_dec,
                   float* __restrict__ out)
{
  extern __shared__ unsigned char smem_raw[];
  unsigned short* s_w    = (unsigned short*)smem_raw;             // [6][112][136]  Wi/Wh x 3 layers
  unsigned short* s_wdec = s_w    + 6 * NROWS * WSTRIDE;          // [64][136]
  unsigned short* s_emb  = s_wdec + 64 * WSTRIDE;                 // [62][136]
  unsigned short* s_h    = s_emb  + VOCAB * WSTRIDE;              // [2 bufs][3][16][136]
  float*          s_bias = (float*)(s_h + 2 * NLAYERS * MROWS * WSTRIDE); // [3][112]
  float*          s_bdec = s_bias + NLAYERS * NROWS;              // [64]

  const int tid = threadIdx.x;

  // Zero the whole arena (K/N zero-padding everywhere; avoids 0*Inf->NaN in WMMA,
  // and gives h(t=-1) == 0 for both ping-pong buffers).
  const int total_u32 =
      (6 * NROWS * WSTRIDE + 64 * WSTRIDE + VOCAB * WSTRIDE +
       2 * NLAYERS * MROWS * WSTRIDE) / 2 + NLAYERS * NROWS + 64;
  unsigned* z = (unsigned*)smem_raw;
  for (int i = tid; i < total_u32; i += TPB) z[i] = 0u;
  __syncthreads();

  // Stage weights (f32 -> bf16) into LDS, row-major W[n][k] so B-frags are contiguous.
  for (int i = tid; i < NLAYERS * HID * HID; i += TPB) {
    int l = i / (HID * HID), r = i % (HID * HID);
    int n = r / HID, k = r % HID;
    s_w[(l * 2 + 0) * NROWS * WSTRIDE + n * WSTRIDE + k] = f2bf(W_ih[i]);
    s_w[(l * 2 + 1) * NROWS * WSTRIDE + n * WSTRIDE + k] = f2bf(W_hh[i]);
  }
  for (int i = tid; i < VOCAB * HID; i += TPB) {           // W_dec and emb are both [62][100]
    int n = i / HID, k = i % HID;
    s_wdec[n * WSTRIDE + k] = f2bf(W_dec[i]);
    s_emb [n * WSTRIDE + k] = f2bf(emb[i]);
  }
  for (int i = tid; i < NLAYERS * HID; i += TPB) {
    int l = i / HID, n = i % HID;
    s_bias[l * NROWS + n] = b_ih[i] + b_hh[i];
  }
  for (int i = tid; i < VOCAB; i += TPB) s_bdec[i] = b_dec[i];
  __syncthreads();

  const int lane = tid & 31;
  const int wave = tid >> 5;
  const int hi   = lane >> 4;
  const int lr   = lane & 15;
  const int b0   = blockIdx.x * MROWS;

  const int ncol = wave * 16 + lr;        // this wave's hidden output column (<112)

  // Hoist time-invariant weight B-fragments into VGPRs (24 x v16bf = 192 VGPRs).
  v16bf wB[NLAYERS][2][4];
#pragma unroll
  for (int l = 0; l < NLAYERS; ++l)
#pragma unroll
    for (int m = 0; m < 2; ++m)
#pragma unroll
      for (int ks = 0; ks < 4; ++ks)
        wB[l][m][ks] = load_bfrag(s_w + (l * 2 + m) * NROWS * WSTRIDE, ncol, ks * 32 + hi * 16);

  v16bf wD[4];
#pragma unroll
  for (int ks = 0; ks < 4; ++ks)
    wD[ks] = load_bfrag(s_wdec, (wave & 3) * 16 + lr, ks * 32 + hi * 16);

  float bias_l[NLAYERS];
#pragma unroll
  for (int l = 0; l < NLAYERS; ++l) bias_l[l] = s_bias[l * NROWS + ncol];
  const float bdv = s_bdec[(wave & 3) * 16 + lr];

  const long long logitsN = (long long)BATCH * SEQ * VOCAB;
  float* hiddenOut = out + logitsN;       // [NLAYERS][BATCH][HID]

  const int* idRow = ids + (b0 + lr) * SEQ;
  int id = idRow[0];                      // software-pipelined token id

  for (int t = 0; t < SEQ; ++t) {
    const unsigned short* aRow0 = s_emb + id * WSTRIDE;
    if (t + 1 < SEQ) id = idRow[t + 1];   // prefetch next id (hidden behind compute)

    const int bufW = t & 1;               // write h(t) here
    const int bufR = bufW ^ 1;            // read  h(t-1) here

#pragma unroll
    for (int l = 0; l < NLAYERS; ++l) {
      const unsigned short* xRow =
          (l == 0) ? aRow0
                   : (s_h + (bufW * NLAYERS + (l - 1)) * MROWS * WSTRIDE + lr * WSTRIDE);
      const unsigned short* hRow =
          s_h + (bufR * NLAYERS + l) * MROWS * WSTRIDE + lr * WSTRIDE;

      // Batch all A-fragment LDS loads ahead of the WMMA chain so they pipeline.
      v16bf ax[4], ah[4];
#pragma unroll
      for (int ks = 0; ks < 4; ++ks) {
        ax[ks] = load_afrag(xRow, ks * 32, hi);
        ah[ks] = load_afrag(hRow, ks * 32, hi);
      }

      v8f c;
#pragma unroll
      for (int r = 0; r < 8; ++r) c[r] = bias_l[l];     // b_ih + b_hh broadcast
#pragma unroll
      for (int ks = 0; ks < 4; ++ks) c = wmma_bf16(ax[ks], wB[l][0][ks], c);  // x @ Wi^T
#pragma unroll
      for (int ks = 0; ks < 4; ++ks) c = wmma_bf16(ah[ks], wB[l][1][ks], c);  // h @ Wh^T

      float hv[8];
#pragma unroll
      for (int r = 0; r < 8; ++r) hv[r] = fast_tanh(c[r]);

      if (ncol < HID) {
        unsigned short* hw = s_h + (bufW * NLAYERS + l) * MROWS * WSTRIDE + ncol;
#pragma unroll
        for (int r = 0; r < 8; ++r) hw[(r + hi * 8) * WSTRIDE] = f2bf(hv[r]);
        if (t == SEQ - 1) {                // final hidden state -> output tail
          float* hd = hiddenOut + ((long long)l * BATCH + b0) * HID + ncol;
#pragma unroll
          for (int r = 0; r < 8; ++r) hd[(r + hi * 8) * HID] = hv[r];
        }
      }
      __syncthreads();                     // h_l(t) visible before layer l+1 reads it
    }

    // Decoder: logits[b,t,:] = h2 @ Wdec^T + bdec  (V=62 padded to 4 tiles of 16)
    {
      const unsigned short* hRow =
          s_h + (bufW * NLAYERS + 2) * MROWS * WSTRIDE + lr * WSTRIDE;
      v16bf ad[4];
#pragma unroll
      for (int ks = 0; ks < 4; ++ks) ad[ks] = load_afrag(hRow, ks * 32, hi);
      v8f c;
#pragma unroll
      for (int r = 0; r < 8; ++r) c[r] = bdv;
#pragma unroll
      for (int ks = 0; ks < 4; ++ks) c = wmma_bf16(ad[ks], wD[ks], c);
      const int vcol = wave * 16 + lr;
      if (wave < 4 && vcol < VOCAB) {
#pragma unroll
        for (int r = 0; r < 8; ++r) {
          long long b = b0 + r + hi * 8;
          out[(b * SEQ + t) * VOCAB + vcol] = c[r];
        }
      }
    }
  }
}

extern "C" void kernel_launch(void* const* d_in, const int* in_sizes, int n_in,
                              void* d_out, int out_size, void* d_ws, size_t ws_size,
                              hipStream_t stream) {
  (void)in_sizes; (void)n_in; (void)d_ws; (void)ws_size; (void)out_size;
  const int*   ids   = (const int*)  d_in[0];
  const float* emb   = (const float*)d_in[1];
  const float* W_ih  = (const float*)d_in[2];
  const float* W_hh  = (const float*)d_in[3];
  const float* b_ih  = (const float*)d_in[4];
  const float* b_hh  = (const float*)d_in[5];
  const float* W_dec = (const float*)d_in[6];
  const float* b_dec = (const float*)d_in[7];
  float* out = (float*)d_out;

  const size_t shmem =
      (size_t)(6 * NROWS * WSTRIDE + 64 * WSTRIDE + VOCAB * WSTRIDE +
               2 * NLAYERS * MROWS * WSTRIDE) * sizeof(unsigned short) +
      (size_t)(NLAYERS * NROWS + 64) * sizeof(float);

  hipFuncSetAttribute((const void*)charrnn_fused,
                      hipFuncAttributeMaxDynamicSharedMemorySize, (int)shmem);

  dim3 grid(BATCH / MROWS);   // 32 workgroups x 16 batch rows
  dim3 block(TPB);            // 7 waves: one 16-col hidden tile per wave
  charrnn_fused<<<grid, block, shmem, stream>>>(ids, emb, W_ih, W_hh, b_ih, b_hh,
                                                W_dec, b_dec, out);
}
